// TransformerEncoder_2379411882571
// MI455X (gfx1250) — matrix operand order repaired and checked by reading, not compiled, not run on hardware
//
#include <hip/hip_runtime.h>
#include <hip/hip_bf16.h>

// ---------------------------------------------------------------------------
// Types for CDNA5 WMMA (wave32): D(16x16,f32) = A(16x32,bf16) x B(32x16,bf16) + C
// ---------------------------------------------------------------------------
typedef __attribute__((ext_vector_type(16))) __bf16 v16bf;
typedef __attribute__((ext_vector_type(8)))  __bf16 v8bf;
typedef __attribute__((ext_vector_type(8)))  float  v8f;
typedef __attribute__((ext_vector_type(8)))  float  v8fv;
typedef __attribute__((ext_vector_type(16))) float  v16fv;

#define WMMA_BF16(A, Bv, C) \
  __builtin_amdgcn_wmma_f32_16x16x32_bf16(false, (A), false, (Bv), (short)0, (C), false, false)

// Model constants
#define BN   16
#define SN   2048
#define DN   128
#define HN   4
#define DH   32
#define FFN  512
#define LN_  4
#define OUTN 30
#define MROWS (BN * SN)       // 32768
#define D3   (3 * DN)         // 384

// Vector f32 -> bf16 fragment builders (lower to v_cvt_pk_bf16_f32, no movs)
__device__ __forceinline__ v16bf cvt2x8(v8fv a, v8fv b) {
  v8bf lo = __builtin_convertvector(a, v8bf);
  v8bf hi = __builtin_convertvector(b, v8bf);
  return __builtin_shufflevector(lo, hi, 0, 1, 2, 3, 4, 5, 6, 7,
                                         8, 9, 10, 11, 12, 13, 14, 15);
}
__device__ __forceinline__ v16bf cvt16(v16fv a) {
  return __builtin_convertvector(a, v16bf);
}

// ---------------------------------------------------------------------------
// fp32 -> bf16 weight conversion
// ---------------------------------------------------------------------------
__global__ void convert_f32_bf16_kernel(const float* __restrict__ src,
                                        __bf16* __restrict__ dst, int n) {
  int i = blockIdx.x * 256 + threadIdx.x;
  if (i < n) dst[i] = (__bf16)src[i];
}

// ---------------------------------------------------------------------------
// GEMM: C[M,N] = A[M,K](f32, cvt->bf16) @ W[N,K](bf16)^T + bias, optional ReLU
// K is a template constant => fully unrolled, software-pipelined K loop.
// Wave: 16(M) x 64(N) macro-tile (A fragment reused over 4 WMMAs per K-step).
// Block: 256 threads = 8 waves => 128 x 64 tile. Grid: (N/64, M/128).
// ---------------------------------------------------------------------------
template <int KT, bool RELU>
__global__ __launch_bounds__(256)
void gemm_bf16_kernel(const float* __restrict__ A, const __bf16* __restrict__ W,
                      const float* __restrict__ bias, float* __restrict__ C,
                      int N) {
  const int lane  = threadIdx.x & 31;
  const int wave  = threadIdx.x >> 5;
  const int half  = lane >> 4;     // 0: lanes 0-15, 1: lanes 16-31
  const int l16   = lane & 15;
  const int tileM = blockIdx.y * 128 + wave * 16;
  const int tileN = blockIdx.x * 64;

  v8f acc[4] = {};

  const int row = tileM + l16;               // A-frag: row per lane%16
  const float* Arow = A + (size_t)row * KT;

#pragma unroll
  for (int kk = 0; kk < KT; kk += 32) {
    // Prefetch next K-slab of A (speculative; dropped if OOB)
    __builtin_prefetch(Arow + kk + 64, 0, 0);

    // A fragment (contiguous K layout: lane<16 -> K {k..k+15},
    // lane>=16 -> K {k+16..k+31})
    const v8fv f1 = *(const v8fv*)(Arow + kk + (half << 4));
    const v8fv f2 = *(const v8fv*)(Arow + kk + (half << 4) + 8);
    const v16bf a = cvt2x8(f1, f2);

#pragma unroll
    for (int i = 0; i < 4; ++i) {
      // B fragment = W^T: lane holds col n = tileN+16i+l16,
      // K {0..15} (lanes 0-15) / {16..31} (lanes 16-31): contiguous 16 bf16.
      const __bf16* bp =
          W + (size_t)(tileN + i * 16 + l16) * KT + kk + (half << 4);
      acc[i] = WMMA_BF16(a, *(const v16bf*)bp, acc[i]);
    }
  }

  // C layout: VGPR j -> row j (+8 for upper half-lanes), col = lane%16
#pragma unroll
  for (int i = 0; i < 4; ++i) {
    const int col = tileN + i * 16 + l16;
    const float bv = bias[col];
#pragma unroll
    for (int j = 0; j < 8; ++j) {
      const int r = tileM + j + (half << 3);
      float v = acc[i][j] + bv;
      if (RELU) v = fmaxf(v, 0.0f);
      C[(size_t)r * N + col] = v;
    }
  }
}

// ---------------------------------------------------------------------------
// Flash attention (dh=32 == WMMA K!). One wave owns 16 query rows; streams
// keys/values in chunks of 32 with online softmax. P re-laid C->A via LDS,
// V staged transposed in LDS. Both staged as fp32 (direct b32 stores from
// load/acc VGPRs, no mov_b16 repacking); bf16 conversion happens on the
// contiguous vector-load side when assembling fragments.
// Block: 128 threads = 4 waves (64 query rows). Grid: (S/64, B*H).
// ---------------------------------------------------------------------------
__global__ __launch_bounds__(128)
void flash_attn_kernel(const float* __restrict__ qkv, float* __restrict__ out) {
  const int lane = threadIdx.x & 31;
  const int wave = threadIdx.x >> 5;
  const int half = lane >> 4;
  const int l16  = lane & 15;

  const int bh = blockIdx.y;
  const int b  = bh >> 2;      // / HN
  const int h  = bh & 3;       // % HN
  const size_t rowBase = (size_t)b * SN;
  const int qRow0 = blockIdx.x * 64 + wave * 16;

  const int qc = h * DH;             // q cols in qkv row
  const int kc = DN + h * DH;        // k cols
  const int vc = 2 * DN + h * DH;    // v cols

  __shared__ alignas(64) float ldsP[4][16 * 32];   // P (fp32)  2KB/wave
  __shared__ alignas(64) float ldsV[4][32 * 32];   // V^T (fp32) 4KB/wave
  float* P  = ldsP[wave];
  float* Vt = ldsV[wave];

  const float scale = 0.17677669529663687f;  // 1/sqrt(32), folded into Q

  // Q A-fragment (loaded once, pre-scaled)
  v16bf aQ;
  {
    const float* qp = qkv + (rowBase + qRow0 + l16) * D3 + qc + (half << 4);
    const v8fv q1 = *(const v8fv*)qp * scale;
    const v8fv q2 = *(const v8fv*)(qp + 8) * scale;
    aQ = cvt2x8(q1, q2);
  }

  v8f o0 = {}, o1 = {};
  float m[8], l[8];
#pragma unroll
  for (int j = 0; j < 8; ++j) { m[j] = -3.0e38f; l[j] = 0.0f; }

  for (int kb = 0; kb < SN; kb += 32) {
    // ---- stage V chunk transposed (fp32): Vt[dh][key_local] ----
    {
      const float* vp = qkv + (rowBase + kb + lane) * D3 + vc;
      const v16fv va = *(const v16fv*)vp;
      const v16fv vb = *(const v16fv*)(vp + 16);
#pragma unroll
      for (int d = 0; d < 16; ++d) {
        Vt[d * 32 + lane]        = va[d];
        Vt[(16 + d) * 32 + lane] = vb[d];
      }
    }

    // ---- K^T B-fragments (K-dim = dh = 32, N = 16 keys each) ----
    const float* kp0 = qkv + (rowBase + kb + l16) * D3 + kc + (half << 4);
    const float* kp1 = qkv + (rowBase + kb + 16 + l16) * D3 + kc + (half << 4);
    const v16bf bK0 = cvt16(*(const v16fv*)kp0);
    const v16bf bK1 = cvt16(*(const v16fv*)kp1);

    v8f s0 = {}, s1 = {};
    s0 = WMMA_BF16(aQ, bK0, s0);
    s1 = WMMA_BF16(aQ, bK1, s1);

    // ---- online softmax update (row stats via 16-lane-half shuffles) ----
    float corr[8];
#pragma unroll
    for (int j = 0; j < 8; ++j) {
      float cm = fmaxf(s0[j], s1[j]);
      cm = fmaxf(cm, __shfl_xor(cm, 1, 32));
      cm = fmaxf(cm, __shfl_xor(cm, 2, 32));
      cm = fmaxf(cm, __shfl_xor(cm, 4, 32));
      cm = fmaxf(cm, __shfl_xor(cm, 8, 32));
      const float mn = fmaxf(m[j], cm);
      corr[j] = __expf(m[j] - mn);
      m[j] = mn;
      const float p0 = __expf(s0[j] - mn);
      const float p1 = __expf(s1[j] - mn);
      s0[j] = p0; s1[j] = p1;
      float rs = p0 + p1;
      rs += __shfl_xor(rs, 1, 32);
      rs += __shfl_xor(rs, 2, 32);
      rs += __shfl_xor(rs, 4, 32);
      rs += __shfl_xor(rs, 8, 32);
      l[j] = l[j] * corr[j] + rs;
      o0[j] *= corr[j];
      o1[j] *= corr[j];
    }

    // ---- P: C-layout regs -> LDS (fp32 b32 stores) -> A-fragment layout ----
#pragma unroll
    for (int j = 0; j < 8; ++j) {
      const int r = j + (half << 3);
      P[r * 32 + l16]      = s0[j];
      P[r * 32 + 16 + l16] = s1[j];
    }
    __syncthreads();

    const float* pp = P + l16 * 32 + (half << 4);
    const v16bf aP = cvt2x8(*(const v8fv*)pp, *(const v8fv*)(pp + 8));

    // V B-fragments from transposed LDS (contiguous vector loads + pk cvt)
    const v16bf bV0 = cvt16(*(const v16fv*)(Vt + l16 * 32 + (half << 4)));
    const v16bf bV1 = cvt16(*(const v16fv*)(Vt + (16 + l16) * 32 + (half << 4)));

    o0 = WMMA_BF16(aP, bV0, o0);
    o1 = WMMA_BF16(aP, bV1, o1);
    __syncthreads();  // protect LDS reuse next chunk
  }

  // ---- epilogue: normalize and store (B,S,D) with head offset ----
#pragma unroll
  for (int j = 0; j < 8; ++j) {
    const float inv = 1.0f / l[j];
    const int r = qRow0 + j + (half << 3);
    float* op = out + (rowBase + r) * DN + h * DH;
    op[l16]      = o0[j] * inv;
    op[16 + l16] = o1[j] * inv;
  }
}

// ---------------------------------------------------------------------------
// Fused residual + LayerNorm over D=128. One wave per row (4 cols/lane).
// Block 256 = 8 rows. In-place safe (each element read once by its lane).
// ---------------------------------------------------------------------------
__global__ __launch_bounds__(256)
void ln_residual_kernel(const float* __restrict__ R, const float* __restrict__ Y,
                        const float* __restrict__ g, const float* __restrict__ bt,
                        float* __restrict__ O) {
  const int lane = threadIdx.x & 31;
  const int wave = threadIdx.x >> 5;
  const size_t row = (size_t)blockIdx.x * 8 + wave;

  float4 a = ((const float4*)(R + row * DN))[lane];
  float4 y = ((const float4*)(Y + row * DN))[lane];
  float x0 = a.x + y.x, x1 = a.y + y.y, x2 = a.z + y.z, x3 = a.w + y.w;

  float s = x0 + x1 + x2 + x3;
  s += __shfl_xor(s, 16, 32); s += __shfl_xor(s, 8, 32);
  s += __shfl_xor(s, 4, 32);  s += __shfl_xor(s, 2, 32);
  s += __shfl_xor(s, 1, 32);
  const float mean = s * (1.0f / 128.0f);

  const float d0 = x0 - mean, d1 = x1 - mean, d2 = x2 - mean, d3 = x3 - mean;
  float v = d0 * d0 + d1 * d1 + d2 * d2 + d3 * d3;
  v += __shfl_xor(v, 16, 32); v += __shfl_xor(v, 8, 32);
  v += __shfl_xor(v, 4, 32);  v += __shfl_xor(v, 2, 32);
  v += __shfl_xor(v, 1, 32);
  const float rstd = rsqrtf(v * (1.0f / 128.0f) + 1e-5f);

  const int c = lane * 4;
  float4 o;
  o.x = d0 * rstd * g[c + 0] + bt[c + 0];
  o.y = d1 * rstd * g[c + 1] + bt[c + 1];
  o.z = d2 * rstd * g[c + 2] + bt[c + 2];
  o.w = d3 * rstd * g[c + 3] + bt[c + 3];
  ((float4*)(O + row * DN))[lane] = o;
}

// ---------------------------------------------------------------------------
// Head: logits (lane==class), softmax over 30 lanes, then the exact
// 16-iteration rebalance scan. One wave per batch row; block = 16 waves.
// ---------------------------------------------------------------------------
__device__ __forceinline__ float wsum32(float x) {
  x += __shfl_xor(x, 16, 32); x += __shfl_xor(x, 8, 32);
  x += __shfl_xor(x, 4, 32);  x += __shfl_xor(x, 2, 32);
  x += __shfl_xor(x, 1, 32);
  return x;
}

__global__ __launch_bounds__(512)
void head_kernel(const float* __restrict__ Hf, const float* __restrict__ fcw,
                 const float* __restrict__ fcb, float* __restrict__ outp) {
  const int lane = threadIdx.x & 31;
  const int b    = threadIdx.x >> 5;
  const float* xr = Hf + ((size_t)b * SN + (SN - 1)) * DN;

  float logit = -3.0e38f;
  if (lane < OUTN) {
    float s = fcb[lane];
    const float* w = fcw + lane * DN;
#pragma unroll 8
    for (int k = 0; k < DN; ++k) s += xr[k] * w[k];
    logit = s;
  }
  float mx = logit;
  mx = fmaxf(mx, __shfl_xor(mx, 16, 32)); mx = fmaxf(mx, __shfl_xor(mx, 8, 32));
  mx = fmaxf(mx, __shfl_xor(mx, 4, 32));  mx = fmaxf(mx, __shfl_xor(mx, 2, 32));
  mx = fmaxf(mx, __shfl_xor(mx, 1, 32));
  const float e = (lane < OUTN) ? __expf(logit - mx) : 0.0f;
  const float se = wsum32(e);
  const float w0 = e / se;

  // _rebalance scan (LB=0.0, UB=0.3, ITERS=16), replicated exactly
  float oldv = w0;
  float wc   = fminf(fmaxf(w0, 0.0f), 0.3f);
  float res  = w0;
  bool done  = false;
#pragma unroll
  for (int it = 0; it < 16; ++it) {
    const float leftover = wsum32((lane < OUTN) ? (oldv - wc) : 0.0f);
    const float noms = (lane < OUTN && wc != 0.3f) ? wc : 0.0f;
    float denom = wsum32(noms);
    denom = (denom == 0.0f) ? 1.0f : denom;
    const float cand = wc + leftover * noms / denom;
    res = done ? res : cand;
    const float anyv = wsum32((lane < OUTN && cand > 0.3f) ? 1.0f : 0.0f);
    done = done || (anyv == 0.0f);
    oldv = done ? oldv : cand;
    wc   = done ? wc : fminf(fmaxf(cand, 0.0f), 0.3f);
  }
  if (lane < OUTN) outp[b * OUTN + lane] = res;
}

// ---------------------------------------------------------------------------
// Orchestration
// ---------------------------------------------------------------------------
extern "C" void kernel_launch(void* const* d_in, const int* in_sizes, int n_in,
                              void* d_out, int out_size, void* d_ws, size_t ws_size,
                              hipStream_t stream) {
  const float* x         = (const float*)d_in[0];
  const float* in_proj_w = (const float*)d_in[1];
  const float* in_proj_b = (const float*)d_in[2];
  const float* out_w     = (const float*)d_in[3];
  const float* out_b     = (const float*)d_in[4];
  const float* ln1_g     = (const float*)d_in[5];
  const float* ln1_b     = (const float*)d_in[6];
  const float* ff1_w     = (const float*)d_in[7];
  const float* ff1_b     = (const float*)d_in[8];
  const float* ff2_w     = (const float*)d_in[9];
  const float* ff2_b     = (const float*)d_in[10];
  const float* ln2_g     = (const float*)d_in[11];
  const float* ln2_b     = (const float*)d_in[12];
  const float* fc_w      = (const float*)d_in[13];
  const float* fc_b      = (const float*)d_in[14];

  char* ws = (char*)d_ws;
  // bf16 weight region
  __bf16* wq = (__bf16*)(ws + 0);          // 4*384*128
  __bf16* wo = (__bf16*)(ws + 393216);     // 4*128*128
  __bf16* w1 = (__bf16*)(ws + 524288);     // 4*512*128
  __bf16* w2 = (__bf16*)(ws + 1048576);    // 4*128*512
  // f32 activation buffers
  float* bufH   = (float*)(ws + 1572864);   // 32768*128  (current hidden, in-place LN)
  float* bufT1  = (float*)(ws + 18350080);  // 32768*128  (attn out)
  float* bufQKV = (float*)(ws + 35127296);  // 32768*384  (qkv / proj / f2 reuse)
  float* bufFF  = (float*)(ws + 85458944);  // 32768*512  (ff1 out)

  const int NWQ = LN_ * D3 * DN, NWO = LN_ * DN * DN;
  const int NW1 = LN_ * FFN * DN, NW2 = LN_ * DN * FFN;
  convert_f32_bf16_kernel<<<(NWQ + 255) / 256, 256, 0, stream>>>(in_proj_w, wq, NWQ);
  convert_f32_bf16_kernel<<<(NWO + 255) / 256, 256, 0, stream>>>(out_w, wo, NWO);
  convert_f32_bf16_kernel<<<(NW1 + 255) / 256, 256, 0, stream>>>(ff1_w, w1, NW1);
  convert_f32_bf16_kernel<<<(NW2 + 255) / 256, 256, 0, stream>>>(ff2_w, w2, NW2);

  hipMemcpyAsync(bufH, x, (size_t)MROWS * DN * sizeof(float),
                 hipMemcpyDeviceToDevice, stream);

  for (int l = 0; l < LN_; ++l) {
    // QKV projection: (32768,384) = H @ Wqkv^T + b
    gemm_bf16_kernel<DN, false><<<dim3(D3 / 64, MROWS / 128), 256, 0, stream>>>(
        bufH, wq + (size_t)l * D3 * DN, in_proj_b + l * D3, bufQKV, D3);
    // Flash attention -> bufT1 (32768,128)
    flash_attn_kernel<<<dim3(SN / 64, BN * HN), 128, 0, stream>>>(bufQKV, bufT1);
    // Output projection -> bufQKV (reused as (32768,128))
    gemm_bf16_kernel<DN, false><<<dim3(DN / 64, MROWS / 128), 256, 0, stream>>>(
        bufT1, wo + (size_t)l * DN * DN, out_b + l * DN, bufQKV, DN);
    // h = LN(h + attn_proj)  (in place over bufH)
    ln_residual_kernel<<<MROWS / 8, 256, 0, stream>>>(
        bufH, bufQKV, ln1_g + l * DN, ln1_b + l * DN, bufH);
    // FF1 + ReLU -> bufFF (32768,512)
    gemm_bf16_kernel<DN, true><<<dim3(FFN / 64, MROWS / 128), 256, 0, stream>>>(
        bufH, w1 + (size_t)l * FFN * DN, ff1_b + l * FFN, bufFF, FFN);
    // FF2 -> bufQKV (32768,128), K=512
    gemm_bf16_kernel<FFN, false><<<dim3(DN / 64, MROWS / 128), 256, 0, stream>>>(
        bufFF, w2 + (size_t)l * DN * FFN, ff2_b + l * DN, bufQKV, DN);
    // h = LN(h + ff)  (in place)
    ln_residual_kernel<<<MROWS / 8, 256, 0, stream>>>(
        bufH, bufQKV, ln2_g + l * DN, ln2_b + l * DN, bufH);
  }

  head_kernel<<<1, 512, 0, stream>>>(bufH, fc_w, fc_b, (float*)d_out);
}
